// SampleAttention_43662637532097
// MI455X (gfx1250) — matrix operand order repaired and checked by reading, not compiled, not run on hardware
//
#include <hip/hip_runtime.h>
#include <hip/hip_bf16.h>

// ---------------------------------------------------------------------------
// Causal multi-head attention forward, MI455X (gfx1250), wave32 + WMMA f16.
// T=4096, C=768, H=12, D=64.
// Pass 0: one-shot f32->f16 conversion of x and weights (memory-bound)
// Pass 1: QKV projection, pure f16 WMMA GEMM, 32x64 tile per wave
// Pass 2: flash attention, 16 queries per wave, online softmax
// Pass 3: output projection, pure f16 WMMA GEMM, f32 store
// ---------------------------------------------------------------------------

#define T_SEQ  4096
#define C_EMB  768
#define N_HEADS 12
#define HEAD_D 64

typedef __attribute__((ext_vector_type(16))) _Float16 v16h;
typedef __attribute__((ext_vector_type(8)))  _Float16 v8h;
typedef __attribute__((ext_vector_type(8)))  float    v8f;

static __device__ inline v8f wmma_f16(v16h a, v16h b, v8f c) {
  // 8 args: (neg_a, A, neg_b, B, c_mod, C, reuse_a, reuse_b)
  return __builtin_amdgcn_wmma_f32_16x16x32_f16(false, a, false, b, (short)0, c,
                                                false, false);
}

static __device__ inline v8f zero8() {
  v8f z;
#pragma unroll
  for (int i = 0; i < 8; ++i) z[i] = 0.0f;
  return z;
}

static __device__ inline v16h cat16(v8h lo, v8h hi) {
  return __builtin_shufflevector(lo, hi, 0, 1, 2, 3, 4, 5, 6, 7,
                                 8, 9, 10, 11, 12, 13, 14, 15);
}

// A-matrix fragment (16x32 f16) from a row-major f16 source.
// Lane l<16: row l, K = 0..15; lane l>=16: row l-16, K = 16..31.
static __device__ inline v16h load_a_f16(const _Float16* __restrict__ src, int ld,
                                         int row, int kbase, int hs) {
  const _Float16* p = src + (size_t)row * ld + kbase + (hs ? 16 : 0);
  v8h lo = *(const v8h*)(p);
  v8h hi = *(const v8h*)(p + 8);
  return cat16(lo, hi);
}

// B-matrix fragment (32x16 f16): lane n<16 holds column n, K=0..15 contiguous;
// lane n>=16 holds column n-16, K=16..31.  `p` points at the 16 contiguous
// halves for this lane (row of the K-major / transposed operand).
static __device__ inline v16h load_b_f16(const _Float16* __restrict__ p) {
  v8h lo = *(const v8h*)(p);
  v8h hi = *(const v8h*)(p + 8);
  return cat16(lo, hi);
}

static __device__ inline v16h load_b_f16_rm(const _Float16* __restrict__ src,
                                            int ld, int row, int kbase, int hs) {
  return load_b_f16(src + (size_t)row * ld + kbase + 16 * hs);
}

// ---------------------------------------------------------------------------
// Pass 0: bulk f32 -> f16 conversion (8 elements per thread).
// ---------------------------------------------------------------------------
__global__ __launch_bounds__(256) void cvt_kernel(const float* __restrict__ src,
                                                  _Float16* __restrict__ dst,
                                                  int n8) {
  const int idx = blockIdx.x * 256 + threadIdx.x;
  if (idx >= n8) return;
  const float4 u0 = *(const float4*)(src + (size_t)idx * 8);
  const float4 u1 = *(const float4*)(src + (size_t)idx * 8 + 4);
  union { v8h v; _Float16 h[8]; } r;
  r.h[0] = (_Float16)u0.x; r.h[1] = (_Float16)u0.y;
  r.h[2] = (_Float16)u0.z; r.h[3] = (_Float16)u0.w;
  r.h[4] = (_Float16)u1.x; r.h[5] = (_Float16)u1.y;
  r.h[6] = (_Float16)u1.z; r.h[7] = (_Float16)u1.w;
  *(v8h*)(dst + (size_t)idx * 8) = r.v;
}

// ---------------------------------------------------------------------------
// Pass 1: fused QKV projection.  y = x @ W^T  (W stored [out,in], f16).
// blockIdx.z selects {Q,K,V}; blockIdx.y = head; each of 8 waves computes a
// 32(M) x 64(N) tile -> block covers 256 tokens x one head.
// Q scaled by 1/sqrt(64); V stored transposed per head: vt[h][d][t].
// ---------------------------------------------------------------------------
__global__ __launch_bounds__(256) void qkv_kernel(
    const _Float16* __restrict__ xh, const _Float16* __restrict__ wqh,
    const _Float16* __restrict__ wkh, const _Float16* __restrict__ wvh,
    _Float16* __restrict__ qh, _Float16* __restrict__ kh,
    _Float16* __restrict__ vth) {
  const int lane = threadIdx.x & 31;
  const int wave = threadIdx.x >> 5;
  const int l16 = lane & 15;
  const int hs = lane >> 4;
  const int m0 = blockIdx.x * 256 + wave * 32;      // token tile (32 rows)
  const int h = blockIdx.y;                         // head
  const int n0 = h * 64;                            // output-channel tile
  const int which = blockIdx.z;
  const _Float16* w = (which == 0) ? wqh : (which == 1) ? wkh : wvh;

  v8f acc[2][4];
#pragma unroll
  for (int mi = 0; mi < 2; ++mi)
#pragma unroll
    for (int j = 0; j < 4; ++j) acc[mi][j] = zero8();

  for (int c = 0; c < C_EMB; c += 32) {
    const v16h a0 = load_a_f16(xh, C_EMB, m0 + l16, c, hs);
    const v16h a1 = load_a_f16(xh, C_EMB, m0 + 16 + l16, c, hs);
#pragma unroll
    for (int j = 0; j < 4; ++j) {
      const v16h b = load_b_f16_rm(w, C_EMB, n0 + 16 * j + l16, c, hs);
      acc[0][j] = wmma_f16(a0, b, acc[0][j]);
      acc[1][j] = wmma_f16(a1, b, acc[1][j]);
    }
  }

  if (which == 2) {
    // V transposed: vt[h][d][t]; C/D element i -> row t (contiguous in i)
#pragma unroll
    for (int mi = 0; mi < 2; ++mi)
#pragma unroll
      for (int j = 0; j < 4; ++j) {
        const int d = 16 * j + l16;
        _Float16* dst = vth + ((size_t)h * 64 + d) * T_SEQ + m0 + 16 * mi + 8 * hs;
#pragma unroll
        for (int i = 0; i < 8; ++i) dst[i] = (_Float16)acc[mi][j][i];
      }
  } else {
    _Float16* dstbase = (which == 0) ? qh : kh;
    const float scale = (which == 0) ? 0.125f : 1.0f;  // fold 1/sqrt(D) into Q
#pragma unroll
    for (int mi = 0; mi < 2; ++mi)
#pragma unroll
      for (int j = 0; j < 4; ++j) {
        const int d = 16 * j + l16;
#pragma unroll
        for (int i = 0; i < 8; ++i) {
          const int t = m0 + 16 * mi + i + 8 * hs;
          dstbase[((size_t)h * T_SEQ + t) * 64 + d] =
              (_Float16)(acc[mi][j][i] * scale);
        }
      }
  }
}

// ---------------------------------------------------------------------------
// Pass 2: causal flash attention.  Each wave owns 16 query rows of one head.
// Key blocks of 32; online softmax in f32; P recast to f16 via per-wave LDS
// transpose (C-layout -> A-layout).  O written to ao[t][h*64+d] (f16).
// ---------------------------------------------------------------------------
__global__ __launch_bounds__(256) void attn_kernel(
    const _Float16* __restrict__ qh, const _Float16* __restrict__ kh,
    const _Float16* __restrict__ vth, _Float16* __restrict__ ao) {
  __shared__ _Float16 lds_p[8][16][32];  // per-wave 16x32 P tile (1 KB each)

  const int lane = threadIdx.x & 31;
  const int wave = threadIdx.x >> 5;
  const int l16 = lane & 15;
  const int hs = lane >> 4;
  const int h = blockIdx.y;
  const int q0 = blockIdx.x * 128 + wave * 16;

  const _Float16* Q = qh + (size_t)h * T_SEQ * 64;
  const _Float16* K = kh + (size_t)h * T_SEQ * 64;
  const _Float16* Vt = vth + (size_t)h * 64 * T_SEQ;

  // Q fragments for K-dim 0..31 and 32..63 (loaded once).
  const v16h qa0 = load_a_f16(Q, 64, q0 + l16, 0, hs);
  const v16h qa1 = load_a_f16(Q, 64, q0 + l16, 32, hs);

  v8f oacc[4];
#pragma unroll
  for (int f = 0; f < 4; ++f) oacc[f] = zero8();
  float m_i[8], l_i[8];
#pragma unroll
  for (int i = 0; i < 8; ++i) { m_i[i] = -INFINITY; l_i[i] = 0.0f; }

  for (int s0 = 0; s0 <= q0 + 15; s0 += 32) {
    // ---- S = Q @ K^T for this 16x32 key block (two 16-col subtiles) ----
    v8f sv0 = zero8(), sv1 = zero8();
    {
      v16h b;
      b = load_b_f16(K + (size_t)(s0 + l16) * 64 + 0 + 16 * hs);
      sv0 = wmma_f16(qa0, b, sv0);
      b = load_b_f16(K + (size_t)(s0 + l16) * 64 + 32 + 16 * hs);
      sv0 = wmma_f16(qa1, b, sv0);
      b = load_b_f16(K + (size_t)(s0 + 16 + l16) * 64 + 0 + 16 * hs);
      sv1 = wmma_f16(qa0, b, sv1);
      b = load_b_f16(K + (size_t)(s0 + 16 + l16) * 64 + 32 + 16 * hs);
      sv1 = wmma_f16(qa1, b, sv1);
    }

    // ---- causal mask (only blocks straddling the diagonal) ----
    if (s0 + 31 > q0) {
#pragma unroll
      for (int i = 0; i < 8; ++i) {
        const int qrow = q0 + i + 8 * hs;
        if (s0 + l16 > qrow)      sv0[i] = -INFINITY;
        if (s0 + 16 + l16 > qrow) sv1[i] = -INFINITY;
      }
    }

    // ---- online softmax: row stats replicated within each 16-lane half ----
    float rmax[8];
#pragma unroll
    for (int i = 0; i < 8; ++i) rmax[i] = fmaxf(sv0[i], sv1[i]);
#pragma unroll
    for (int off = 8; off > 0; off >>= 1) {
#pragma unroll
      for (int i = 0; i < 8; ++i)
        rmax[i] = fmaxf(rmax[i], __shfl_xor(rmax[i], off));
    }

    float corr[8], rsum[8];
#pragma unroll
    for (int i = 0; i < 8; ++i) {
      const float mn = fmaxf(m_i[i], rmax[i]);
      corr[i] = __expf(m_i[i] - mn);
      m_i[i] = mn;
      sv0[i] = __expf(sv0[i] - mn);
      sv1[i] = __expf(sv1[i] - mn);
      rsum[i] = sv0[i] + sv1[i];
    }
#pragma unroll
    for (int off = 8; off > 0; off >>= 1) {
#pragma unroll
      for (int i = 0; i < 8; ++i) rsum[i] += __shfl_xor(rsum[i], off);
    }
#pragma unroll
    for (int i = 0; i < 8; ++i) l_i[i] = l_i[i] * corr[i] + rsum[i];
#pragma unroll
    for (int f = 0; f < 4; ++f)
#pragma unroll
      for (int i = 0; i < 8; ++i) oacc[f][i] *= corr[i];

    // ---- P: C-layout -> A-layout via per-wave LDS bounce ----
    _Float16* pp = &lds_p[wave][0][0];
#pragma unroll
    for (int i = 0; i < 8; ++i) {
      pp[(i + 8 * hs) * 32 + l16] = (_Float16)sv0[i];
      pp[(i + 8 * hs) * 32 + 16 + l16] = (_Float16)sv1[i];
    }
    __asm__ volatile("s_wait_dscnt 0" ::: "memory");  // wave-private RAW on LDS
    const _Float16* pr = &lds_p[wave][l16][hs ? 16 : 0];
    const v8h plo = *(const v8h*)(pr);
    const v8h phi = *(const v8h*)(pr + 8);
    const v16h pa = cat16(plo, phi);

    // ---- O += P @ V (V^T rows are contiguous B fragments) ----
#pragma unroll
    for (int f = 0; f < 4; ++f) {
      const v16h vb =
          load_b_f16(Vt + ((size_t)(16 * f + l16)) * T_SEQ + s0 + 16 * hs);
      oacc[f] = wmma_f16(pa, vb, oacc[f]);
    }
  }

  // ---- epilogue: normalize by l and store f16 attn output [t][h*64+d] ----
#pragma unroll
  for (int f = 0; f < 4; ++f) {
#pragma unroll
    for (int i = 0; i < 8; ++i) {
      const int t = q0 + i + 8 * hs;
      ao[(size_t)t * C_EMB + h * 64 + 16 * f + l16] =
          (_Float16)(oacc[f][i] / l_i[i]);
    }
  }
}

// ---------------------------------------------------------------------------
// Pass 3: output projection  out = attn_out @ wo^T  (f16 operands, f32 out).
// 32(M) x 64(N) tile per wave.
// ---------------------------------------------------------------------------
__global__ __launch_bounds__(256) void proj_kernel(
    const _Float16* __restrict__ ao, const _Float16* __restrict__ woh,
    float* __restrict__ out) {
  const int lane = threadIdx.x & 31;
  const int wave = threadIdx.x >> 5;
  const int l16 = lane & 15;
  const int hs = lane >> 4;
  const int m0 = blockIdx.x * 256 + wave * 32;
  const int n0 = blockIdx.y * 64;

  v8f acc[2][4];
#pragma unroll
  for (int mi = 0; mi < 2; ++mi)
#pragma unroll
    for (int j = 0; j < 4; ++j) acc[mi][j] = zero8();

  for (int c = 0; c < C_EMB; c += 32) {
    const v16h a0 = load_a_f16(ao, C_EMB, m0 + l16, c, hs);
    const v16h a1 = load_a_f16(ao, C_EMB, m0 + 16 + l16, c, hs);
#pragma unroll
    for (int j = 0; j < 4; ++j) {
      const v16h b = load_b_f16_rm(woh, C_EMB, n0 + 16 * j + l16, c, hs);
      acc[0][j] = wmma_f16(a0, b, acc[0][j]);
      acc[1][j] = wmma_f16(a1, b, acc[1][j]);
    }
  }

#pragma unroll
  for (int mi = 0; mi < 2; ++mi)
#pragma unroll
    for (int j = 0; j < 4; ++j) {
#pragma unroll
      for (int i = 0; i < 8; ++i) {
        const int t = m0 + 16 * mi + i + 8 * hs;
        out[(size_t)t * C_EMB + n0 + 16 * j + l16] = acc[mi][j][i];
      }
    }
}

// ---------------------------------------------------------------------------
extern "C" void kernel_launch(void* const* d_in, const int* in_sizes, int n_in,
                              void* d_out, int out_size, void* d_ws,
                              size_t ws_size, hipStream_t stream) {
  const float* x = (const float*)d_in[0];
  const float* wq = (const float*)d_in[1];
  const float* wk = (const float*)d_in[2];
  const float* wv = (const float*)d_in[3];
  const float* wo = (const float*)d_in[4];
  float* out = (float*)d_out;

  // Workspace layout (all f16):
  //   qh | kh | vth  (each H*T*D)  | ao (T*C) | xh (T*C) | wq|wk|wv|wo (C*C)
  const size_t per = (size_t)N_HEADS * T_SEQ * HEAD_D;   // = T*C
  const size_t wsz = (size_t)C_EMB * C_EMB;
  _Float16* qh  = (_Float16*)d_ws;
  _Float16* kh  = qh + per;
  _Float16* vth = kh + per;
  _Float16* ao  = vth + per;
  _Float16* xh  = ao + per;
  _Float16* wqh = xh + per;
  _Float16* wkh = wqh + wsz;
  _Float16* wvh = wkh + wsz;
  _Float16* woh = wvh + wsz;

  // Pass 0: one-shot conversions (memory-bound).
  const int nx8 = (int)(per / 8);
  const int nw8 = (int)(wsz / 8);
  cvt_kernel<<<dim3((nx8 + 255) / 256), 256, 0, stream>>>(x, xh, nx8);
  cvt_kernel<<<dim3((nw8 + 255) / 256), 256, 0, stream>>>(wq, wqh, nw8);
  cvt_kernel<<<dim3((nw8 + 255) / 256), 256, 0, stream>>>(wk, wkh, nw8);
  cvt_kernel<<<dim3((nw8 + 255) / 256), 256, 0, stream>>>(wv, wvh, nw8);
  cvt_kernel<<<dim3((nw8 + 255) / 256), 256, 0, stream>>>(wo, woh, nw8);

  // Pass 1-3.
  qkv_kernel<<<dim3(T_SEQ / 256, N_HEADS, 3), 256, 0, stream>>>(
      xh, wqh, wkh, wvh, qh, kh, vth);
  attn_kernel<<<dim3(T_SEQ / 128, N_HEADS), 256, 0, stream>>>(qh, kh, vth, ao);
  proj_kernel<<<dim3(T_SEQ / 256, C_EMB / 64), 256, 0, stream>>>(ao, woh, out);
}
